// ROIBaseline_1142461300794
// MI455X (gfx1250) — compile-verified, matrix operand order checked by reading, not run on hardware
//
#include <hip/hip_runtime.h>
#include <math.h>

// ---------------- problem constants ----------------
#define BATCH   1024
#define NROI    333
#define ROIS    60          // in-features per ROI
#define RED     16          // out-features per ROI
#define CONCAT  (NROI*RED)  // 5328
#define KP      5344        // CONCAT padded to multiple of 32 for f16 WMMA
#define H1      1024
#define H2      300
#define NCLS    180
#define NCLSP   192         // padded to 16
#define EPSBN   1e-5f
#define SLOPE   0.3f

typedef float    v2f  __attribute__((ext_vector_type(2)));
typedef float    v8f  __attribute__((ext_vector_type(8)));
typedef _Float16 hv8  __attribute__((ext_vector_type(8)));
typedef _Float16 hv16 __attribute__((ext_vector_type(16)));
typedef int      v4i_gcc __attribute__((__vector_size__(16)));

// ---- fp32 WMMA: D = A(16x4)*B(4x16) + C ----
__device__ __forceinline__ v8f wmma_f32(v2f a, v2f b, v8f c) {
  return __builtin_amdgcn_wmma_f32_16x16x4_f32(false, a, false, b, (short)0, c,
                                               false, false);
}

// ---- f16 WMMA: D = A(16x32)*B(32x16) + C ----
__device__ __forceinline__ v8f wmma_f16(hv16 a, hv16 b, v8f c) {
  return __builtin_amdgcn_wmma_f32_16x16x32_f16(false, a, false, b, (short)0, c,
                                                false, false);
}

// compose a 16-half fragment from two 16B segments at p and p+16 halves
__device__ __forceinline__ hv16 frag16(const _Float16* p) {
  hv8 x0 = *(const hv8*)p;
  hv8 x1 = *(const hv8*)(p + 16);
  return __builtin_shufflevector(x0, x1, 0, 1, 2, 3, 4, 5, 6, 7,
                                 8, 9, 10, 11, 12, 13, 14, 15);
}

// ---- async global->LDS (CDNA5), guarded so compile can't fail ----
#if __has_builtin(__builtin_amdgcn_global_load_async_to_lds_b128) && \
    __has_builtin(__builtin_amdgcn_s_wait_asynccnt)
#define USE_ASYNC_LDS 1
#else
#define USE_ASYNC_LDS 0
#endif

__device__ __forceinline__ void cp16(_Float16* dst, const _Float16* src) {
#if USE_ASYNC_LDS
  // builtin signature (from hipcc diagnostic): (v4i AS1*, v4i AS3*, imm offset, cpol)
  // low 32 bits of a flat shared address are the LDS byte offset (ISA aperture rule)
  __builtin_amdgcn_global_load_async_to_lds_b128(
      (__attribute__((address_space(1))) v4i_gcc*)(unsigned long long)(const void*)src,
      (__attribute__((address_space(3))) v4i_gcc*)(unsigned int)(unsigned long long)(void*)dst,
      0, 0);
#else
  *(hv8*)dst = *(const hv8*)src;
#endif
}

__device__ __forceinline__ void cp_wait() {
#if USE_ASYNC_LDS
  __builtin_amdgcn_s_wait_asynccnt(0);
#endif
}

// ---------------- 1) grouped encoder GEMM (fp32 WMMA, K=60) ----------------
__global__ void roi_enc_gemm(const float* __restrict__ x,
                             const float* __restrict__ W,
                             const float* __restrict__ bias,
                             float* __restrict__ h) {
  const int mt   = blockIdx.x;
  const int r    = blockIdx.y;
  const int lane = threadIdx.x;
  const int half = lane >> 4;
  const int l15  = lane & 15;
  const int m0   = mt * 16;

  const float* xr = x + (size_t)(m0 + l15) * (NROI * ROIS) + (size_t)r * ROIS;
  const float* wr = W + ((size_t)r * RED + l15) * ROIS;

  v8f acc = {};
  #pragma unroll
  for (int k = 0; k < ROIS; k += 4) {
    const int kk = k + 2 * half;
    v2f a; a.x = xr[kk]; a.y = xr[kk + 1];
    v2f b; b.x = wr[kk]; b.y = wr[kk + 1];
    acc = wmma_f32(a, b, acc);
  }

  const int   col = r * RED + l15;
  const float bs  = bias[col];
  #pragma unroll
  for (int j = 0; j < 8; ++j) {
    const int row = m0 + j + 8 * half;
    h[(size_t)row * CONCAT + col] = acc[j] + bs;
  }
}

// ---------------- 2) BN batch statistics over axis 0 ----------------
__global__ void roi_bn_stats(const float* __restrict__ h,
                             float* __restrict__ mean,
                             float* __restrict__ rstd, int ncols) {
  const int tx = threadIdx.x & 63;
  const int ty = threadIdx.x >> 6;
  const int c  = blockIdx.x * 64 + tx;
  float s = 0.f, s2 = 0.f;
  if (c < ncols) {
    for (int b = ty; b < BATCH; b += 4) {
      const float v = h[(size_t)b * ncols + c];
      s += v; s2 += v * v;
    }
  }
  __shared__ float sh[4][64], sh2[4][64];
  sh[ty][tx] = s; sh2[ty][tx] = s2;
  __syncthreads();
  if (ty == 0 && c < ncols) {
    const float S1 = sh[0][tx] + sh[1][tx] + sh[2][tx] + sh[3][tx];
    const float S2 = sh2[0][tx] + sh2[1][tx] + sh2[2][tx] + sh2[3][tx];
    const float m  = S1 * (1.f / BATCH);
    const float var = S2 * (1.f / BATCH) - m * m;
    mean[c] = m;
    rstd[c] = rsqrtf(var + EPSBN);
  }
}

// ---------------- 3a) BN apply + LeakyReLU + f16 hi/lo split (c path) ----------------
__global__ void roi_bn_apply_split(const float* __restrict__ h,
                                   const float* __restrict__ mean,
                                   const float* __restrict__ rstd,
                                   const float* __restrict__ gamma,
                                   const float* __restrict__ beta,
                                   _Float16* __restrict__ chi,
                                   _Float16* __restrict__ clo) {
  const int col = blockIdx.x * 256 + threadIdx.x;
  const int row = blockIdx.y;
  if (col >= KP) return;
  float v = 0.f;
  if (col < CONCAT) {
    float t = (h[(size_t)row * CONCAT + col] - mean[col]) * rstd[col] * gamma[col]
              + beta[col];
    v = t >= 0.f ? t : SLOPE * t;
  }
  const _Float16 hi = (_Float16)v;
  chi[(size_t)row * KP + col] = hi;
  clo[(size_t)row * KP + col] = (_Float16)(v - (float)hi);
}

// ---------------- 3b) pad + f16 hi/lo split of a weight matrix ----------------
__global__ void roi_split_pad(const float* __restrict__ w,
                              _Float16* __restrict__ whi,
                              _Float16* __restrict__ wlo, int scols) {
  const int col = blockIdx.x * 256 + threadIdx.x;
  const int row = blockIdx.y;
  if (col >= KP) return;
  const float v = (col < scols) ? w[(size_t)row * scols + col] : 0.f;
  const _Float16 hi = (_Float16)v;
  whi[(size_t)row * KP + col] = hi;
  wlo[(size_t)row * KP + col] = (_Float16)(v - (float)hi);
}

// ---------------- 3c) BN apply + LeakyReLU, fp32 in place (z path) ----------------
__global__ void roi_bn_apply(float* __restrict__ h,
                             const float* __restrict__ mean,
                             const float* __restrict__ rstd,
                             const float* __restrict__ gamma,
                             const float* __restrict__ beta, int ncols) {
  const size_t i = (size_t)blockIdx.x * 256 + threadIdx.x;
  if (i >= (size_t)BATCH * ncols) return;
  const int c = (int)(i % ncols);
  float v = (h[i] - mean[c]) * rstd[c] * gamma[c] + beta[c];
  h[i] = v >= 0.f ? v : SLOPE * v;
}

// ---------------- 4) main GEMM via split-f16 (f16x3): z = c @ w1^T + b1 ----------------
// M=N=1024, K=5344. 64x64 macro tile / 128 threads (4 waves), wave = 32x32
// (4 accumulators: 12 WMMAs per 16 ds_load_b128 per K-chunk of 32).
// Double-buffered LDS; global->LDS via async DMA when available.
#define LDH 40   // LDS row stride in halves (80B: 16B-aligned, conflict-spread)
__global__ __launch_bounds__(128) void roi_gemm1_f16x3(
    const _Float16* __restrict__ Ahg, const _Float16* __restrict__ Alg,
    const _Float16* __restrict__ Bhg, const _Float16* __restrict__ Blg,
    const float* __restrict__ b1, float* __restrict__ z) {
  __shared__ _Float16 Ah[2][64 * LDH], Al[2][64 * LDH];
  __shared__ _Float16 Bh[2][64 * LDH], Bl[2][64 * LDH];

  const int tid  = threadIdx.x;
  const int wv   = tid >> 5;
  const int lane = tid & 31;
  const int half = lane >> 4;
  const int l15  = lane & 15;
  const int msub = wv & 1;            // 2 m-subtiles of 32
  const int nsub = wv >> 1;           // 2 n-subtiles of 32
  const int m0   = blockIdx.y * 64;
  const int n0   = blockIdx.x * 64;

  const int grow = tid >> 1;          // 0..63
  const int gs0  = (tid & 1) * 16;    // 0 or 16 (halves); each thread does 2x16B/array

  v8f acc00 = {}, acc01 = {}, acc10 = {}, acc11 = {};

  auto stage = [&](int buf, int k0) {
    const size_t arow = (size_t)(m0 + grow) * KP + k0;
    const size_t brow = (size_t)(n0 + grow) * KP + k0;
    const int ldst = grow * LDH;
    cp16(&Ah[buf][ldst + gs0],     Ahg + arow + gs0);
    cp16(&Ah[buf][ldst + gs0 + 8], Ahg + arow + gs0 + 8);
    cp16(&Al[buf][ldst + gs0],     Alg + arow + gs0);
    cp16(&Al[buf][ldst + gs0 + 8], Alg + arow + gs0 + 8);
    cp16(&Bh[buf][ldst + gs0],     Bhg + brow + gs0);
    cp16(&Bh[buf][ldst + gs0 + 8], Bhg + brow + gs0 + 8);
    cp16(&Bl[buf][ldst + gs0],     Blg + brow + gs0);
    cp16(&Bl[buf][ldst + gs0 + 8], Blg + brow + gs0 + 8);
  };

  stage(0, 0);
  cp_wait();
  __syncthreads();

  int cur = 0;
  for (int k0 = 0; k0 < KP; k0 += 32) {
    const int nxt = cur ^ 1;
    if (k0 + 32 < KP) stage(nxt, k0 + 32);   // DMA next chunk while computing

    const int ar0 = (msub * 32 + l15) * LDH + 8 * half;
    const int ar1 = ar0 + 16 * LDH;
    const int br0 = (nsub * 32 + l15) * LDH + 8 * half;
    const int br1 = br0 + 16 * LDH;

    const hv16 aH0 = frag16(&Ah[cur][ar0]);
    const hv16 aH1 = frag16(&Ah[cur][ar1]);
    const hv16 aL0 = frag16(&Al[cur][ar0]);
    const hv16 aL1 = frag16(&Al[cur][ar1]);
    const hv16 bH0 = frag16(&Bh[cur][br0]);
    const hv16 bH1 = frag16(&Bh[cur][br1]);
    const hv16 bL0 = frag16(&Bl[cur][br0]);
    const hv16 bL1 = frag16(&Bl[cur][br1]);

    acc00 = wmma_f16(aH0, bL0, acc00);
    acc00 = wmma_f16(aL0, bH0, acc00);
    acc00 = wmma_f16(aH0, bH0, acc00);
    acc01 = wmma_f16(aH0, bL1, acc01);
    acc01 = wmma_f16(aL0, bH1, acc01);
    acc01 = wmma_f16(aH0, bH1, acc01);
    acc10 = wmma_f16(aH1, bL0, acc10);
    acc10 = wmma_f16(aL1, bH0, acc10);
    acc10 = wmma_f16(aH1, bH0, acc10);
    acc11 = wmma_f16(aH1, bL1, acc11);
    acc11 = wmma_f16(aL1, bH1, acc11);
    acc11 = wmma_f16(aH1, bH1, acc11);

    cp_wait();
    __syncthreads();
    cur = nxt;
  }

  const int rb = m0 + msub * 32;
  const int c0 = n0 + nsub * 32 + l15;
  const float bi0 = b1[c0], bi1 = b1[c0 + 16];
  #pragma unroll
  for (int j = 0; j < 8; ++j) {
    const int row0 = rb + j + 8 * half;
    const int row1 = row0 + 16;
    z[(size_t)row0 * H1 + c0]      = acc00[j] + bi0;
    z[(size_t)row0 * H1 + c0 + 16] = acc01[j] + bi1;
    z[(size_t)row1 * H1 + c0]      = acc10[j] + bi0;
    z[(size_t)row1 * H1 + c0 + 16] = acc11[j] + bi1;
  }
}

// ---------------- 5) GEMM2: reg = z @ w2^T + b2  (N=300, fp32 WMMA) ----------------
__global__ void roi_gemm2(const float* __restrict__ z,
                          const float* __restrict__ w2,
                          const float* __restrict__ b2,
                          float* __restrict__ reg) {
  const int mt = blockIdx.x, nt = blockIdx.y;
  const int lane = threadIdx.x, half = lane >> 4, l15 = lane & 15;
  const int m0 = mt * 16;
  const int n  = nt * 16 + l15;
  const float bscale = (n < H2) ? 1.f : 0.f;
  const int nclamp = (n < H2) ? n : (H2 - 1);
  const float* zr = z  + (size_t)(m0 + l15) * H1;
  const float* wr = w2 + (size_t)nclamp * H1;

  v8f acc = {};
  for (int k = 0; k < H1; k += 4) {
    const int kk = k + 2 * half;
    v2f a; a.x = zr[kk];          a.y = zr[kk + 1];
    v2f b; b.x = wr[kk] * bscale; b.y = wr[kk + 1] * bscale;
    acc = wmma_f32(a, b, acc);
  }
  if (n < H2) {
    const float bs = b2[n];
    #pragma unroll
    for (int j = 0; j < 8; ++j) {
      const int row = m0 + j + 8 * half;
      reg[(size_t)row * H2 + n] = acc[j] + bs;
    }
  }
}

// ---------------- 6) GEMM3: logits = reg @ wc^T + bc  (N=180, K=300) ----------------
__global__ void roi_gemm3(const float* __restrict__ reg,
                          const float* __restrict__ wc,
                          const float* __restrict__ bc,
                          float* __restrict__ logits) {
  const int mt = blockIdx.x, nt = blockIdx.y;
  const int lane = threadIdx.x, half = lane >> 4, l15 = lane & 15;
  const int m0 = mt * 16;
  const int n  = nt * 16 + l15;
  const float bscale = (n < NCLS) ? 1.f : 0.f;
  const int nclamp = (n < NCLS) ? n : (NCLS - 1);
  const float* ar = reg + (size_t)(m0 + l15) * H2;
  const float* wr = wc  + (size_t)nclamp * H2;

  v8f acc = {};
  for (int k = 0; k < H2; k += 4) {
    const int kk = k + 2 * half;
    v2f a; a.x = ar[kk];          a.y = ar[kk + 1];
    v2f b; b.x = wr[kk] * bscale; b.y = wr[kk + 1] * bscale;
    acc = wmma_f32(a, b, acc);
  }
  if (n < NCLS) {
    const float bs = bc[n];
    #pragma unroll
    for (int j = 0; j < 8; ++j) {
      const int row = m0 + j + 8 * half;
      logits[(size_t)row * NCLSP + n] = acc[j] + bs;
    }
  }
}

// ---------------- 7) row softmax over 180 classes ----------------
__global__ void roi_softmax(const float* __restrict__ logits,
                            float* __restrict__ y) {
  const int row = blockIdx.x;
  const int t   = threadIdx.x;       // 64 threads
  __shared__ float sh[64];
  const float* lr = logits + (size_t)row * NCLSP;

  float mx = -3.402823466e38f;
  for (int c = t; c < NCLS; c += 64) mx = fmaxf(mx, lr[c]);
  sh[t] = mx; __syncthreads();
  for (int off = 32; off > 0; off >>= 1) {
    if (t < off) sh[t] = fmaxf(sh[t], sh[t + off]);
    __syncthreads();
  }
  mx = sh[0]; __syncthreads();

  float s = 0.f;
  for (int c = t; c < NCLS; c += 64) s += expf(lr[c] - mx);
  sh[t] = s; __syncthreads();
  for (int off = 32; off > 0; off >>= 1) {
    if (t < off) sh[t] += sh[t + off];
    __syncthreads();
  }
  const float inv = 1.f / sh[0];
  for (int c = t; c < NCLS; c += 64)
    y[(size_t)row * NCLS + c] = expf(lr[c] - mx) * inv;
}

// ---------------- launcher ----------------
extern "C" void kernel_launch(void* const* d_in, const int* in_sizes, int n_in,
                              void* d_out, int out_size, void* d_ws, size_t ws_size,
                              hipStream_t stream) {
  const float* x         = (const float*)d_in[0];
  const float* enc_W     = (const float*)d_in[1];
  const float* enc_b     = (const float*)d_in[2];
  const float* enc_gamma = (const float*)d_in[3];
  const float* enc_beta  = (const float*)d_in[4];
  const float* w1        = (const float*)d_in[5];
  const float* b1        = (const float*)d_in[6];
  const float* g1        = (const float*)d_in[7];
  const float* be1       = (const float*)d_in[8];
  const float* w2        = (const float*)d_in[9];
  const float* b2        = (const float*)d_in[10];
  const float* wc        = (const float*)d_in[11];
  const float* bc        = (const float*)d_in[12];

  float* out     = (float*)d_out;
  float* reg_out = out;                              // [1024,300]
  float* y_pred  = out + (size_t)BATCH * H2;         // [1024,180]

  // workspace: h | mean1 | rstd1 | z | mean2 | rstd2 | logits | chi | clo | w1hi | w1lo
  float* ws     = (float*)d_ws;
  float* h      = ws;
  float* mean1  = h + (size_t)BATCH * CONCAT;
  float* rstd1  = mean1 + CONCAT;
  float* z      = rstd1 + CONCAT;
  float* mean2  = z + (size_t)BATCH * H1;
  float* rstd2  = mean2 + H1;
  float* logits = rstd2 + H1;
  _Float16* chi  = (_Float16*)(logits + (size_t)BATCH * NCLSP);
  _Float16* clo  = chi  + (size_t)BATCH * KP;
  _Float16* w1hi = clo  + (size_t)BATCH * KP;
  _Float16* w1lo = w1hi + (size_t)H1 * KP;

  const int colBlocks = (KP + 255) / 256;

  roi_enc_gemm<<<dim3(BATCH / 16, NROI), 32, 0, stream>>>(x, enc_W, enc_b, h);
  roi_bn_stats<<<(CONCAT + 63) / 64, 256, 0, stream>>>(h, mean1, rstd1, CONCAT);
  roi_bn_apply_split<<<dim3(colBlocks, BATCH), 256, 0, stream>>>(
      h, mean1, rstd1, enc_gamma, enc_beta, chi, clo);
  roi_split_pad<<<dim3(colBlocks, H1), 256, 0, stream>>>(w1, w1hi, w1lo, CONCAT);
  roi_gemm1_f16x3<<<dim3(H1 / 64, BATCH / 64), 128, 0, stream>>>(
      chi, clo, w1hi, w1lo, b1, z);
  roi_bn_stats<<<H1 / 64, 256, 0, stream>>>(z, mean2, rstd2, H1);
  roi_bn_apply<<<(int)(((size_t)BATCH * H1 + 255) / 256), 256, 0, stream>>>(
      z, mean2, rstd2, g1, be1, H1);
  roi_gemm2<<<dim3(BATCH / 16, (H2 + 15) / 16), 32, 0, stream>>>(z, w2, b2, reg_out);
  roi_gemm3<<<dim3(BATCH / 16, NCLSP / 16), 32, 0, stream>>>(reg_out, wc, bc, logits);
  roi_softmax<<<BATCH, 64, 0, stream>>>(logits, y_pred);
}
